// xLSTM_70428873719994
// MI455X (gfx1250) — matrix operand order repaired and checked by reading, not compiled, not run on hardware
//
#include <hip/hip_runtime.h>
#include <hip/hip_bf16.h>
#include <math.h>

// ---------------------------------------------------------------------------
// xLSTM block for MI455X (gfx1250, wave32).  GEMMs run on
// v_wmma_f32_16x16x32_bf16 fed from double-buffered LDS tiles filled with
// async global->LDS copies; scan/LN/conv/gates are VALU kernels.
// ---------------------------------------------------------------------------

#define BN   8
#define SN   1024
#define DN   1024
#define HN   8
#define HDN  64
#define HIDN 512
#define PN   2048
#define BSN  (BN * SN)          // 8192 rows

typedef __bf16 bf16_t;
typedef bf16_t v16bf __attribute__((ext_vector_type(16)));
typedef bf16_t v8bf  __attribute__((ext_vector_type(8)));
typedef float  v8f   __attribute__((ext_vector_type(8)));
typedef int    v4i   __attribute__((ext_vector_type(4)));

union FragU { v16bf v; v8bf h[2]; };

#if defined(__has_builtin)
#if __has_builtin(__builtin_amdgcn_global_load_async_to_lds_b128) && \
    __has_builtin(__builtin_amdgcn_s_wait_asynccnt)
#define ASYNC_CP 1
#endif
#endif
#ifndef ASYNC_CP
#define ASYNC_CP 0
#endif

// builtin signature (probe-confirmed): (v4i as1* src, v4i as3* dst, imm off, imm cpol)
#define GASP(p) ((__attribute__((address_space(1))) v4i*)(p))
#define LASP(p) ((__attribute__((address_space(3))) v4i*)(p))

__device__ __forceinline__ unsigned short f2bf(float f) {
  union { float f; unsigned u; } cv; cv.f = f;
  unsigned r = cv.u + 0x7FFFu + ((cv.u >> 16) & 1u);   // round-to-nearest-even
  return (unsigned short)(r >> 16);
}
__device__ __forceinline__ float bf2f(unsigned short h) {
  union { unsigned u; float f; } cv; cv.u = ((unsigned)h) << 16; return cv.f;
}

// ---------------------------------------------------------------------------
// 1) Row LayerNorm (D=1024) -> bf16
// ---------------------------------------------------------------------------
__global__ __launch_bounds__(256)
void layernorm_rows(const float* __restrict__ x, const float* __restrict__ g,
                    const float* __restrict__ b, unsigned short* __restrict__ out) {
  int row = blockIdx.x;
  const float* xr = x + (size_t)row * DN;
  float s = 0.f, s2 = 0.f;
  for (int j = threadIdx.x; j < DN; j += 256) { float v = xr[j]; s += v; s2 += v * v; }
  #pragma unroll
  for (int off = 16; off; off >>= 1) { s += __shfl_xor(s, off); s2 += __shfl_xor(s2, off); }
  __shared__ float rs[8], rs2[8];
  if ((threadIdx.x & 31) == 0) { rs[threadIdx.x >> 5] = s; rs2[threadIdx.x >> 5] = s2; }
  __syncthreads();
  float ts = 0.f, ts2 = 0.f;
  #pragma unroll
  for (int i = 0; i < 8; i++) { ts += rs[i]; ts2 += rs2[i]; }
  float mu  = ts / (float)DN;
  float var = ts2 / (float)DN - mu * mu;
  float inv = rsqrtf(var + 1e-6f);
  unsigned short* orow = out + (size_t)row * DN;
  for (int j = threadIdx.x; j < DN; j += 256)
    orow[j] = f2bf((xr[j] - mu) * inv * g[j] + b[j]);
}

// ---------------------------------------------------------------------------
// 2) fp32 -> bf16 weight conversion
// ---------------------------------------------------------------------------
__global__ __launch_bounds__(256)
void cvt_bf16(const float* __restrict__ in, unsigned short* __restrict__ out, long n) {
  long i = (long)blockIdx.x * 256 + threadIdx.x;
  if (i < n) out[i] = f2bf(in[i]);
}

// ---------------------------------------------------------------------------
// 3) WMMA bf16 GEMM:  C(M,N) = A(M,K) * W(N,K)^T + bias, fused epilogue.
//    Block = 256 threads = 8 waves; tile = 128(M) x 64(N); K step 32.
//    A/B tiles staged in double-buffered LDS via async global->LDS copies.
//    ep: 0 = none, 1 = *0.125 (k-scale), 2 = sigmoid (o-gate)
// ---------------------------------------------------------------------------
#define LSTRIDE 40   // bf16 elements per LDS row (32 data + 8 pad, 16B aligned, bank-spread)

__global__ __launch_bounds__(256)
void gemm_wmma_bf16(const unsigned short* __restrict__ A,
                    const unsigned short* __restrict__ W,
                    const float* __restrict__ bias,
                    const float* __restrict__ resid,
                    float* __restrict__ C,
                    int M, int N, int Kd, int ep) {
  __shared__ unsigned short lA[2][128 * LSTRIDE];
  __shared__ unsigned short lB[2][64 * LSTRIDE];

  int tid  = threadIdx.x;
  int wave = tid >> 5;
  int lane = tid & 31;
  int m0 = blockIdx.x * 128;
  int n0 = blockIdx.y * 64;

  // tile-loader assignment: 16B (8 x bf16) chunks
  int arow0 = tid >> 2;              // rows 0..63 of A tile
  int arow1 = arow0 + 64;            // rows 64..127
  int acol  = (tid & 3) * 8;
  int brow  = tid >> 2;              // rows 0..63 of B tile
  int bcol  = (tid & 3) * 8;

  auto issue = [&](int buf, int kk) {
    const unsigned short* ga0 = A + (size_t)(m0 + arow0) * Kd + kk + acol;
    const unsigned short* ga1 = A + (size_t)(m0 + arow1) * Kd + kk + acol;
    const unsigned short* gb  = W + (size_t)(n0 + brow) * Kd + kk + bcol;
    unsigned short* la0 = &lA[buf][arow0 * LSTRIDE + acol];
    unsigned short* la1 = &lA[buf][arow1 * LSTRIDE + acol];
    unsigned short* lb  = &lB[buf][brow * LSTRIDE + bcol];
#if ASYNC_CP
    __builtin_amdgcn_global_load_async_to_lds_b128(GASP(ga0), LASP(la0), 0, 0);
    __builtin_amdgcn_global_load_async_to_lds_b128(GASP(ga1), LASP(la1), 0, 0);
    __builtin_amdgcn_global_load_async_to_lds_b128(GASP(gb),  LASP(lb),  0, 0);
#else
    *(v8bf*)la0 = *(const v8bf*)ga0;
    *(v8bf*)la1 = *(const v8bf*)ga1;
    *(v8bf*)lb  = *(const v8bf*)gb;
#endif
  };

  // fragment addressing
  int frow  = wave * 16 + (lane & 15);   // A row within 128-row tile
  int ahalf = (lane >> 4) * 8;           // A: lanes 0-15 -> K{0..7,16..23}; 16-31 -> {8..15,24..31}
  int bhalf = (lane >> 4) * 8;           // B: lanes 0-15 -> K{0..7,16..23}; 16-31 -> {8..15,24..31}

  v8f acc[4];
  #pragma unroll
  for (int j = 0; j < 4; j++) acc[j] = (v8f)0.0f;

  int nk = Kd >> 5;
  issue(0, 0);
  for (int ik = 0; ik < nk; ik++) {
    int buf = ik & 1;
#if ASYNC_CP
    __builtin_amdgcn_s_wait_asynccnt(0);   // own writes for this buffer done
#endif
    __syncthreads();                        // tile published; everyone done reading buf^1
    if (ik + 1 < nk) issue(buf ^ 1, (ik + 1) << 5);  // overlaps with WMMA below

    // batch ALL fragment ds_loads before the WMMA chain: one DScnt drain,
    // then 4 back-to-back v_wmma ops (better DS pipelining + XDL co-exec)
    FragU ua, ub[4];
    const unsigned short* la = &lA[buf][frow * LSTRIDE + ahalf];
    ua.h[0] = *(const v8bf*)(la);
    ua.h[1] = *(const v8bf*)(la + 16);
    #pragma unroll
    for (int j = 0; j < 4; j++) {
      const unsigned short* lb = &lB[buf][(j * 16 + (lane & 15)) * LSTRIDE + bhalf];
      ub[j].h[0] = *(const v8bf*)(lb);
      ub[j].h[1] = *(const v8bf*)(lb + 16);
    }
    #pragma unroll
    for (int j = 0; j < 4; j++) {
      acc[j] = __builtin_amdgcn_wmma_f32_16x16x32_bf16(
          false, ua.v, false, ub[j].v, (short)0, acc[j], false, false);
    }
  }

  // C layout: lanes 0-15 -> N=lane, M=r ; lanes 16-31 -> N=lane-16, M=r+8
  int mbase = m0 + wave * 16 + ((lane >> 4) ? 8 : 0);
  #pragma unroll
  for (int j = 0; j < 4; j++) {
    int n = n0 + j * 16 + (lane & 15);
    float bv = bias ? bias[n] : 0.f;
    #pragma unroll
    for (int r = 0; r < 8; r++) {
      float v = acc[j][r] + bv;
      if (ep == 1)      v *= 0.125f;
      else if (ep == 2) v = 1.f / (1.f + __expf(-v));
      size_t idx = (size_t)(mbase + r) * (size_t)N + (size_t)n;
      if (resid) v += resid[idx];
      C[idx] = v;
    }
  }
}

// ---------------------------------------------------------------------------
// 4) causal feature-axis conv (K=4) + SiLU -> x_c(bf16); also x_t -> bf16
// ---------------------------------------------------------------------------
__global__ __launch_bounds__(256)
void conv_silu_kernel(const float* __restrict__ xt, const float* __restrict__ w,
                      const float* __restrict__ cb,
                      unsigned short* __restrict__ xc_bf,
                      unsigned short* __restrict__ xt_bf) {
  size_t idx = (size_t)blockIdx.x * 256 + threadIdx.x;
  size_t row = idx >> 11;           // / 2048
  int j = (int)(idx & 2047);
  const float* xr = xt + (row << 11);
  float acc = cb[0];
  #pragma unroll
  for (int m = 0; m < 4; m++) {
    int jj = j - 3 + m;
    if (jj >= 0) acc += w[m] * xr[jj];
  }
  float s = acc / (1.f + __expf(-acc));
  xc_bf[idx] = f2bf(s);
  xt_bf[idx] = f2bf(xr[j]);
}

// ---------------------------------------------------------------------------
// 5) i/f gates: x_c(row,2048) . W_i/W_f (8,2048) + soft cap. One block per row.
// ---------------------------------------------------------------------------
__global__ __launch_bounds__(256)
void gates_kernel(const unsigned short* __restrict__ xc,
                  const float* __restrict__ Wi, const float* __restrict__ bi,
                  const float* __restrict__ Wf, const float* __restrict__ bfv,
                  float* __restrict__ ip, float* __restrict__ fp) {
  int row = blockIdx.x;
  const unsigned short* xr = xc + (size_t)row * PN;
  float acc[16];
  #pragma unroll
  for (int h = 0; h < 16; h++) acc[h] = 0.f;
  for (int j = threadIdx.x; j < PN; j += 256) {
    float xv = bf2f(xr[j]);
    #pragma unroll
    for (int h = 0; h < 8; h++) {
      acc[h]     += xv * Wi[h * PN + j];
      acc[8 + h] += xv * Wf[h * PN + j];
    }
  }
  __shared__ float red[16][8];
  #pragma unroll
  for (int h = 0; h < 16; h++) {
    float v = acc[h];
    #pragma unroll
    for (int off = 16; off; off >>= 1) v += __shfl_xor(v, off);
    if ((threadIdx.x & 31) == 0) red[h][threadIdx.x >> 5] = v;
  }
  __syncthreads();
  if (threadIdx.x < 16) {
    float v = 0.f;
    #pragma unroll
    for (int i = 0; i < 8; i++) v += red[threadIdx.x][i];
    v += (threadIdx.x < 8) ? bi[threadIdx.x] : bfv[threadIdx.x - 8];
    v = 15.f * tanhf(v * (1.f / 15.f));
    if (threadIdx.x < 8) ip[(size_t)row * 8 + threadIdx.x] = v;
    else                 fp[(size_t)row * 8 + threadIdx.x - 8] = v;
  }
}

// ---------------------------------------------------------------------------
// 6) mLSTM scan: one block per (b,h); 64x64 state in registers (16 f32/thread)
// ---------------------------------------------------------------------------
__global__ __launch_bounds__(256)
void mlstm_scan(const float* __restrict__ q, const float* __restrict__ k,
                const float* __restrict__ v, const float* __restrict__ ip,
                const float* __restrict__ fp, float* __restrict__ hs) {
  int b = blockIdx.x >> 3, h = blockIdx.x & 7;
  int t = threadIdx.x;
  int d = t >> 2, qd = t & 3, eb = qd << 4;
  float c[16];
  #pragma unroll
  for (int j = 0; j < 16; j++) c[j] = 0.f;
  float nreg = 1.f, m = 0.f;
  __shared__ float lq[64], lk[64], lv[64], lnum[64], lden[8];
  for (int s = 0; s < SN; s++) {
    __syncthreads();
    size_t base = ((size_t)b * SN + s) * HIDN + h * HDN;
    if (t < 64) { lq[t] = q[base + t]; lk[t] = k[base + t]; lv[t] = v[base + t]; }
    __syncthreads();
    size_t gb = ((size_t)b * SN + s) * HN + h;
    float it = ip[gb], ft = fp[gb];
    float mt = fmaxf(ft + m, it);
    float ie = __expf(it - mt);
    float fe = __expf(ft - mt + m);
    float vd = lv[d];
    float num = 0.f;
    #pragma unroll
    for (int j = 0; j < 16; j++) {
      int e = eb + j;
      c[j] = fe * c[j] + ie * vd * lk[e];
      num += c[j] * lq[e];
    }
    num += __shfl_xor(num, 1);
    num += __shfl_xor(num, 2);
    if (qd == 0) lnum[d] = num;
    float dp = 0.f;
    if (t < 64) { nreg = fe * nreg + ie * lk[t]; dp = nreg * lq[t]; }
    #pragma unroll
    for (int off = 16; off; off >>= 1) dp += __shfl_xor(dp, off);
    if ((t & 31) == 0) lden[t >> 5] = dp;
    __syncthreads();
    float den = fmaxf(lden[0] + lden[1], 1.0f);
    if (t < 64) hs[base + t] = lnum[t] / den;
    m = mt;
  }
}

// ---------------------------------------------------------------------------
// 7) h = o*hs, per-head LN, (hn + skip) * silu(r_t) -> bf16. Block = one head.
// ---------------------------------------------------------------------------
__global__ __launch_bounds__(64)
void post_kernel(const float* __restrict__ hs, const float* __restrict__ og,
                 const float* __restrict__ mhg, const float* __restrict__ mhb,
                 const float* __restrict__ skip, const float* __restrict__ rt,
                 unsigned short* __restrict__ outpre) {
  int grp = blockIdx.x;             // bs*8 + h
  int t = threadIdx.x;
  int h = grp & 7;
  size_t base = (size_t)(grp >> 3) * HIDN + (size_t)h * HDN + t;
  float hv = og[base] * hs[base];
  float s = hv, s2 = hv * hv;
  #pragma unroll
  for (int off = 16; off; off >>= 1) { s += __shfl_xor(s, off); s2 += __shfl_xor(s2, off); }
  __shared__ float a[2], a2[2];
  if ((t & 31) == 0) { a[t >> 5] = s; a2[t >> 5] = s2; }
  __syncthreads();
  float ts = a[0] + a[1], ts2 = a2[0] + a2[1];
  float mu  = ts * (1.f / 64.f);
  float var = ts2 * (1.f / 64.f) - mu * mu;
  float hn = (hv - mu) * rsqrtf(var + 1e-6f) * mhg[h * HDN + t] + mhb[h * HDN + t];
  float r = rt[base];
  float sil = r / (1.f + __expf(-r));
  outpre[base] = f2bf((hn + skip[base]) * sil);
}

// ---------------------------------------------------------------------------
extern "C" void kernel_launch(void* const* d_in, const int* in_sizes, int n_in,
                              void* d_out, int out_size, void* d_ws, size_t ws_size,
                              hipStream_t stream) {
  (void)in_sizes; (void)n_in; (void)out_size; (void)ws_size;
  const float* x      = (const float*)d_in[0];
  const float* ln_g   = (const float*)d_in[1];
  const float* ln_b   = (const float*)d_in[2];
  const float* mh_g   = (const float*)d_in[3];
  const float* mh_b   = (const float*)d_in[4];
  const float* W_up_l = (const float*)d_in[5];
  const float* b_up_l = (const float*)d_in[6];
  const float* W_up_r = (const float*)d_in[7];
  const float* b_up_r = (const float*)d_in[8];
  const float* W_down = (const float*)d_in[9];
  const float* b_down = (const float*)d_in[10];
  const float* conv_w = (const float*)d_in[11];
  const float* conv_b = (const float*)d_in[12];
  const float* W_skip = (const float*)d_in[13];
  const float* b_skip = (const float*)d_in[14];
  const float* W_i    = (const float*)d_in[15];
  const float* b_i    = (const float*)d_in[16];
  const float* W_f    = (const float*)d_in[17];
  const float* b_f    = (const float*)d_in[18];
  const float* W_o    = (const float*)d_in[19];
  const float* b_o    = (const float*)d_in[20];
  const float* W_q    = (const float*)d_in[21];
  const float* b_q    = (const float*)d_in[22];
  const float* W_k    = (const float*)d_in[23];
  const float* b_k    = (const float*)d_in[24];
  const float* W_v    = (const float*)d_in[25];
  const float* b_v    = (const float*)d_in[26];

  char* ws = (char*)d_ws;
  size_t off = 0;
  auto carve = [&](size_t bytes) -> char* {
    char* p = ws + off;
    off = (off + bytes + 255) & ~(size_t)255;
    return p;
  };

  unsigned short* xn_bf   = (unsigned short*)carve((size_t)BSN * DN * 2);   // 16 MB (reused for hs later)
  float*          xt_f32  = (float*)carve((size_t)BSN * PN * 4);            // 64 MB (reused for q/k/v/o)
  unsigned short* xt_bf   = (unsigned short*)carve((size_t)BSN * PN * 2);   // 32 MB
  unsigned short* xc_bf   = (unsigned short*)carve((size_t)BSN * PN * 2);   // 32 MB
  float*          rt_f32  = (float*)carve((size_t)BSN * HIDN * 4);          // 16 MB
  float*          sk_f32  = (float*)carve((size_t)BSN * HIDN * 4);          // 16 MB
  float*          ip_f32  = (float*)carve((size_t)BSN * HN * 4);
  float*          fp_f32  = (float*)carve((size_t)BSN * HN * 4);
  unsigned short* op_bf   = (unsigned short*)carve((size_t)BSN * HIDN * 2); // 8 MB
  unsigned short* Wupl_bf = (unsigned short*)carve((size_t)PN * DN * 2);
  unsigned short* Wupr_bf = (unsigned short*)carve((size_t)HIDN * DN * 2);
  unsigned short* Wsk_bf  = (unsigned short*)carve((size_t)HIDN * PN * 2);
  unsigned short* Wq_bf   = (unsigned short*)carve((size_t)HIDN * PN * 2);
  unsigned short* Wk_bf   = (unsigned short*)carve((size_t)HIDN * PN * 2);
  unsigned short* Wv_bf   = (unsigned short*)carve((size_t)HIDN * PN * 2);
  unsigned short* Wo_bf   = (unsigned short*)carve((size_t)HIDN * PN * 2);
  unsigned short* Wdn_bf  = (unsigned short*)carve((size_t)DN * HIDN * 2);

  // Buffer reuse: q/k/v/o overlay xt_f32 after conv consumes it; hs overlays xn_bf.
  float* q_f32 = xt_f32;
  float* k_f32 = xt_f32 + (size_t)BSN * HIDN;
  float* v_f32 = xt_f32 + (size_t)2 * BSN * HIDN;
  float* o_f32 = xt_f32 + (size_t)3 * BSN * HIDN;
  float* hs_f32 = (float*)xn_bf;   // 16 MB region, exactly BSN*HIDN*4

  // 1) LayerNorm -> xn bf16
  layernorm_rows<<<BSN, 256, 0, stream>>>(x, ln_g, ln_b, xn_bf);

  // 2) weight conversions
  auto cvt = [&](const float* src, unsigned short* dst, long n) {
    cvt_bf16<<<(unsigned)((n + 255) / 256), 256, 0, stream>>>(src, dst, n);
  };
  cvt(W_up_l, Wupl_bf, (long)PN * DN);
  cvt(W_up_r, Wupr_bf, (long)HIDN * DN);
  cvt(W_skip, Wsk_bf,  (long)HIDN * PN);
  cvt(W_q,    Wq_bf,   (long)HIDN * PN);
  cvt(W_k,    Wk_bf,   (long)HIDN * PN);
  cvt(W_v,    Wv_bf,   (long)HIDN * PN);
  cvt(W_o,    Wo_bf,   (long)HIDN * PN);
  cvt(W_down, Wdn_bf,  (long)DN * HIDN);

  // 3) up projections
  gemm_wmma_bf16<<<dim3(BSN / 128, PN / 64), 256, 0, stream>>>(
      xn_bf, Wupl_bf, b_up_l, nullptr, xt_f32, BSN, PN, DN, 0);
  gemm_wmma_bf16<<<dim3(BSN / 128, HIDN / 64), 256, 0, stream>>>(
      xn_bf, Wupr_bf, b_up_r, nullptr, rt_f32, BSN, HIDN, DN, 0);

  // 4) causal conv + SiLU (also materializes x_t in bf16)
  conv_silu_kernel<<<(unsigned)(((size_t)BSN * PN) / 256), 256, 0, stream>>>(
      xt_f32, conv_w, conv_b, xc_bf, xt_bf);

  // 5) mid projections (q/k/v/o/skip) — q..o overwrite xt_f32, which is now dead
  gemm_wmma_bf16<<<dim3(BSN / 128, HIDN / 64), 256, 0, stream>>>(
      xc_bf, Wsk_bf, b_skip, nullptr, sk_f32, BSN, HIDN, PN, 0);
  gemm_wmma_bf16<<<dim3(BSN / 128, HIDN / 64), 256, 0, stream>>>(
      xc_bf, Wq_bf, b_q, nullptr, q_f32, BSN, HIDN, PN, 0);
  gemm_wmma_bf16<<<dim3(BSN / 128, HIDN / 64), 256, 0, stream>>>(
      xc_bf, Wk_bf, b_k, nullptr, k_f32, BSN, HIDN, PN, 1);   // *1/sqrt(64)
  gemm_wmma_bf16<<<dim3(BSN / 128, HIDN / 64), 256, 0, stream>>>(
      xt_bf, Wv_bf, b_v, nullptr, v_f32, BSN, HIDN, PN, 0);
  gemm_wmma_bf16<<<dim3(BSN / 128, HIDN / 64), 256, 0, stream>>>(
      xt_bf, Wo_bf, b_o, nullptr, o_f32, BSN, HIDN, PN, 2);   // sigmoid

  // 6) i/f gates with soft cap
  gates_kernel<<<BSN, 256, 0, stream>>>(xc_bf, W_i, b_i, W_f, b_f, ip_f32, fp_f32);

  // 7) sequential mLSTM scan (hs overwrites xn_bf, which is now dead)
  mlstm_scan<<<BN * HN, 256, 0, stream>>>(q_f32, k_f32, v_f32, ip_f32, fp_f32, hs_f32);

  // 8) output gate * scan, per-head LN, skip + silu(r) gating -> bf16
  post_kernel<<<BSN * HN, 64, 0, stream>>>(hs_f32, o_f32, mh_g, mh_b, sk_f32, rt_f32, op_bf);

  // 9) down projection + bias + residual x -> d_out (fp32)
  gemm_wmma_bf16<<<dim3(BSN / 128, DN / 64), 256, 0, stream>>>(
      op_bf, Wdn_bf, b_down, x, (float*)d_out, BSN, DN, HIDN, 0);
}
